// Attention_11957188952717
// MI455X (gfx1250) — matrix operand order repaired and checked by reading, not compiled, not run on hardware
//
#include <hip/hip_runtime.h>
#include <hip/hip_bf16.h>
#include <math.h>

// Problem constants (reference: B=2, H=16, S=2048, D=64, fp32)
#define Bz 2
#define Hh 16
#define Ss 2048
#define Dd 64

typedef __attribute__((ext_vector_type(16))) _Float16 v16h;
typedef __attribute__((ext_vector_type(8)))  _Float16 v8h;
typedef __attribute__((ext_vector_type(4)))  _Float16 v4h;
typedef __attribute__((ext_vector_type(8)))  float    v8f;

#define BM 64   // query rows per block (4 waves x 16)
#define BN 32   // keys per tile
#define PS 40   // padded stride (halves) for P scratch rows
#define VS 40   // padded stride (halves) for V^T tile rows

// 1/sqrt(64) * log2(e): score scaling folded into base-2 softmax
#define KSC 0.18033688011112042f

// max with row_ror:N DPP butterfly step (VALU-only 16-lane reduction,
// co-executes with XDL WMMA; replaces ds_bpermute)
#define ROR_MAX(x, ctrl)                                                     \
    fmaxf((x), __builtin_bit_cast(float, __builtin_amdgcn_update_dpp(        \
        __builtin_bit_cast(int, (x)), __builtin_bit_cast(int, (x)),          \
        (ctrl), 0xf, 0xf, true)))

__global__ __launch_bounds__(128, 2)
void fa_wmma_kernel(const float* __restrict__ q,
                    const float* __restrict__ k,
                    const float* __restrict__ v,
                    const int*   __restrict__ mask,
                    float* __restrict__ out)
{
    // Double-buffered K/V tiles: one barrier per iteration, global loads for
    // tile kt+1 overlap with the WMMA/softmax work on tile kt.
    __shared__ _Float16 Ks[2][BN * Dd];     // K tile, row-major [key][d]
    __shared__ _Float16 Vt[2][Dd * VS];     // V tile, transposed [d][key]
    __shared__ _Float16 Ps[4 * 16 * PS];    // per-wave P scratch (C->A layout)

    const int tid  = threadIdx.x;
    const int wave = tid >> 5;
    const int lane = tid & 31;
    const int lm   = lane & 15;
    const int lhi  = lane >> 4;

    const int bh    = blockIdx.y;               // 0 .. B*H-1
    const int qtile = blockIdx.x;               // 0 .. S/BM-1
    const int qbase = qtile * BM + wave * 16;   // first query row for this wave

    const float* qg = q   + (size_t)bh * Ss * Dd;
    const float* kg = k   + (size_t)bh * Ss * Dd;
    const float* vg = v   + (size_t)bh * Ss * Dd;
    float*       og = out + (size_t)bh * Ss * Dd;

    // this thread's slice of a 32x64 tile, as 4 float4s (fully coalesced)
    const int t_key = (tid * 4) >> 6;           // reused below: idx = tid + i*128
    (void)t_key;

    // ---- Q in WMMA A-fragment layout: 16x32 f16 per fragment, 2 fragments ----
    v16h qa[2];
    {
        const float* qr = qg + (size_t)(qbase + lm) * Dd;
        #pragma unroll
        for (int f = 0; f < 2; ++f) {
            #pragma unroll
            for (int r = 0; r < 8; ++r) {
                int k0 = 32 * f + 2 * r + 16 * lhi;
                qa[f][2 * r]     = (_Float16)qr[k0];
                qa[f][2 * r + 1] = (_Float16)qr[k0 + 1];
            }
        }
    }

    // B-fragment of the "ones column": P x ones -> softmax denominator in col 0.
    v16h vones;
    {
        _Float16 ov = (lm == 0) ? (_Float16)1.0f : (_Float16)0.0f;
        #pragma unroll
        for (int j = 0; j < 16; ++j) vones[j] = ov;
    }

    const v8f zero = {0.f, 0.f, 0.f, 0.f, 0.f, 0.f, 0.f, 0.f};
    v8f oacc[4] = {zero, zero, zero, zero};   // O: 16x64 fp32 accum
    v8f lacc = zero;                          // running denominator (col 0)
    float mrow[8];
    #pragma unroll
    for (int r = 0; r < 8; ++r) mrow[r] = -1.0e30f;  // finite floor: no NaN guards

    const int NT = Ss / BN;                   // 64 key tiles

    float4 kreg[4], vreg[4];                  // staged tile (software pipeline)

    // ---- prologue: stage tile 0 and commit it to LDS buffer 0 ----
    {
        const float4* ksrc = (const float4*)kg;
        const float4* vsrc = (const float4*)vg;
        #pragma unroll
        for (int i = 0; i < 4; ++i) {
            kreg[i] = ksrc[tid + i * 128];
            vreg[i] = vsrc[tid + i * 128];
        }
        #pragma unroll
        for (int i = 0; i < 4; ++i) {
            int idx = tid + i * 128;
            int key = idx >> 4;
            int d4  = (idx & 15) << 2;
            v4h hk = {(_Float16)kreg[i].x, (_Float16)kreg[i].y,
                      (_Float16)kreg[i].z, (_Float16)kreg[i].w};
            *(v4h*)&Ks[0][key * Dd + d4] = hk;
            Vt[0][(d4 + 0) * VS + key] = (_Float16)vreg[i].x;
            Vt[0][(d4 + 1) * VS + key] = (_Float16)vreg[i].y;
            Vt[0][(d4 + 2) * VS + key] = (_Float16)vreg[i].z;
            Vt[0][(d4 + 3) * VS + key] = (_Float16)vreg[i].w;
        }
    }

    for (int kt = 0; kt < NT; ++kt) {
        const int buf = kt & 1;
        __syncthreads();   // publishes Ks/Vt[buf]; orders prior readers of buf^1

        // ---- issue next tile's global loads now; they retire under compute ----
        if (kt + 1 < NT) {
            const float4* ksrc = (const float4*)(kg + (size_t)(kt + 1) * BN * Dd);
            const float4* vsrc = (const float4*)(vg + (size_t)(kt + 1) * BN * Dd);
            #pragma unroll
            for (int i = 0; i < 4; ++i) {
                kreg[i] = ksrc[tid + i * 128];
                vreg[i] = vsrc[tid + i * 128];
            }
        }
        if (kt + 2 < NT) {
            __builtin_prefetch(kg + (size_t)(kt + 2) * BN * Dd + tid * 16, 0, 0);
            __builtin_prefetch(vg + (size_t)(kt + 2) * BN * Dd + tid * 16, 0, 0);
        }

        // ---- S = Q x K^T  (16 rows x 32 keys), 4 WMMAs ----
        v8f sacc[2] = {zero, zero};
        #pragma unroll
        for (int nc = 0; nc < 2; ++nc) {
            #pragma unroll
            for (int ks = 0; ks < 2; ++ks) {
                const v16h kb =
                    *(const v16h*)&Ks[buf][(nc * 16 + lm) * Dd + 32 * ks + 16 * lhi];
                sacc[nc] = __builtin_amdgcn_wmma_f32_16x16x32_f16(
                    false, qa[ks], false, kb, (short)0, sacc[nc], false, false);
            }
        }

        // ---- mask + scale + online softmax (base-2 domain) ----
        const int kcol0 = kt * BN;
        float p0[8], p1[8], alpha[8];
        #pragma unroll
        for (int r = 0; r < 8; ++r) {
            const size_t qrow_g = (size_t)(qbase + r + 8 * lhi);
            int m0 = mask[qrow_g * Ss + kcol0 + lm];
            int m1 = mask[qrow_g * Ss + kcol0 + 16 + lm];
            float s0 = m0 ? sacc[0][r] * KSC : -INFINITY;
            float s1 = m1 ? sacc[1][r] * KSC : -INFINITY;
            // 16-lane max butterfly: VALU DPP row rotations, no LDS traffic
            float x = fmaxf(s0, s1);
            x = ROR_MAX(x, 0x121);  // row_ror:1
            x = ROR_MAX(x, 0x122);  // row_ror:2
            x = ROR_MAX(x, 0x124);  // row_ror:4
            x = ROR_MAX(x, 0x128);  // row_ror:8
            float mnew = fmaxf(mrow[r], x);      // >= -1e30, always finite
            alpha[r] = __builtin_amdgcn_exp2f(mrow[r] - mnew);
            p0[r]    = __builtin_amdgcn_exp2f(s0 - mnew);  // exp2(-inf)=0 when masked
            p1[r]    = __builtin_amdgcn_exp2f(s1 - mnew);
            mrow[r]  = mnew;
        }
        // rescale O and denominator accumulators by alpha (per row)
        #pragma unroll
        for (int r = 0; r < 8; ++r) {
            lacc[r] *= alpha[r];
            #pragma unroll
            for (int ds = 0; ds < 4; ++ds)
                oacc[ds][r] *= alpha[r];
        }

        // ---- convert P (C layout) -> A-fragment via wave-private LDS scratch ----
        _Float16* Ph = &Ps[wave * 16 * PS];
        #pragma unroll
        for (int r = 0; r < 8; ++r) {
            Ph[(r + 8 * lhi) * PS + lm]      = (_Float16)p0[r];
            Ph[(r + 8 * lhi) * PS + 16 + lm] = (_Float16)p1[r];
        }
        asm volatile("s_wait_dscnt 0" ::: "memory");  // wave-local store->load ordering
        v16h pa;
        #pragma unroll
        for (int r = 0; r < 8; ++r) {
            int k0 = 2 * r + 16 * lhi;
            pa[2 * r]     = Ph[lm * PS + k0];
            pa[2 * r + 1] = Ph[lm * PS + k0 + 1];
        }

        // ---- denominator: lacc += P x ones (row sums land in column 0) ----
        lacc = __builtin_amdgcn_wmma_f32_16x16x32_f16(
            false, pa, false, vones, (short)0, lacc, false, false);

        // ---- O += P x V  (16x32 * 32x64), 4 WMMAs ----
        #pragma unroll
        for (int ds = 0; ds < 4; ++ds) {
            const v8h* vp = (const v8h*)&Vt[buf][(ds * 16 + lm) * VS + 16 * lhi];
            v16h vb;
            #pragma unroll
            for (int j = 0; j < 8; ++j) { vb[j] = vp[0][j]; vb[8 + j] = vp[1][j]; }
            oacc[ds] = __builtin_amdgcn_wmma_f32_16x16x32_f16(
                false, pa, false, vb, (short)0, oacc[ds], false, false);
        }

        // ---- commit staged tile kt+1 into the other buffer ----
        // Safe: all readers of buf^1 (iteration kt-1) are before this iteration's
        // barrier; its readers (iteration kt+1) are after the next barrier.
        if (kt + 1 < NT) {
            #pragma unroll
            for (int i = 0; i < 4; ++i) {
                int idx = tid + i * 128;
                int key = idx >> 4;
                int d4  = (idx & 15) << 2;
                v4h hk = {(_Float16)kreg[i].x, (_Float16)kreg[i].y,
                          (_Float16)kreg[i].z, (_Float16)kreg[i].w};
                *(v4h*)&Ks[buf ^ 1][key * Dd + d4] = hk;
                Vt[buf ^ 1][(d4 + 0) * VS + key] = (_Float16)vreg[i].x;
                Vt[buf ^ 1][(d4 + 1) * VS + key] = (_Float16)vreg[i].y;
                Vt[buf ^ 1][(d4 + 2) * VS + key] = (_Float16)vreg[i].z;
                Vt[buf ^ 1][(d4 + 3) * VS + key] = (_Float16)vreg[i].w;
            }
        }
    }

    // ---- normalize and write out ----
    #pragma unroll
    for (int r = 0; r < 8; ++r) {
        // denominator lives in column 0 of each 16-lane half: broadcast it
        float lsum = __shfl(lacc[r], lhi << 4, 32);
        float inv  = (lsum > 0.f) ? 1.0f / lsum : 0.f;
        size_t row = (size_t)(qbase + r + 8 * lhi);
        #pragma unroll
        for (int ds = 0; ds < 4; ++ds)
            og[row * Dd + ds * 16 + lm] = oacc[ds][r] * inv;
    }
}

extern "C" void kernel_launch(void* const* d_in, const int* in_sizes, int n_in,
                              void* d_out, int out_size, void* d_ws, size_t ws_size,
                              hipStream_t stream) {
    const float* q    = (const float*)d_in[0];
    const float* k    = (const float*)d_in[1];
    const float* v    = (const float*)d_in[2];
    const int*   mask = (const int*)d_in[3];
    float* out = (float*)d_out;

    dim3 grid(Ss / BM, Bz * Hh);   // (32 query tiles, 32 batch*heads)
    fa_wmma_kernel<<<grid, 128, 0, stream>>>(q, k, v, mask, out);
}
